// _S4MLikeRegressor_23476291240193
// MI455X (gfx1250) — matrix operand order repaired and checked by reading, not compiled, not run on hardware
//
#include <hip/hip_runtime.h>
#include <hip/hip_bf16.h>

// ---------------------------------------------------------------------------
// S4M-like regressor for MI455X (gfx1250, wave32, WMMA).
//   * prep:  fp32->bf16 weight conversion + state zero-init (workspace).
//   * main:  fused GEMM(u,g) + activations + diagonal scan. Grid = B x 8 time
//            chunks; x chunk staged fp32->bf16 in LDS; per-wave WMMA bf16
//            16x16x32 with hoisted B fragments; branch-free activations that
//            co-execute with the XDL pipe; Horner scan with decay powers;
//            exact cross-chunk combine via atomicAdd with decay^(256*(7-c)).
//   * head:  LayerNorm -> WMMA GEMM(512) -> exact GELU -> WMMA GEMM(256).
// ---------------------------------------------------------------------------

typedef __bf16 bf16;
typedef bf16 v16bf __attribute__((ext_vector_type(16)));
typedef bf16 v8bf  __attribute__((ext_vector_type(8)));
typedef bf16 v4bf  __attribute__((ext_vector_type(4)));
typedef float v8f  __attribute__((ext_vector_type(8)));

#define B_     64
#define T_     2048
#define D_     256
#define S_     512
#define O_     256
#define CHUNK  256
#define NCHUNK (T_ / CHUNK)   // 8
#define LPITCH 272            // 256 + 16 pad (bf16 elems) -> 544B rows, bank-spread
#define HPITCH 528            // 512 + 16 pad

union V16 { v16bf v; v8bf h[2]; };

// B-fragment (32x16 bf16) from a row-major bf16 matrix with leading dim ld.
// Lane n=l%16 holds K = kb..kb+7 and kb+16..kb+23 (kb = k*32 + 8*(l/16),
// folded in by caller), matching the A-fragment layout.
__device__ inline v16bf load_bfrag(const bf16* __restrict__ Wb, int sCol, int ld, int kb) {
  V16 r;
  const bf16* p = Wb + (size_t)sCol * ld + kb;
  r.h[0] = *(const v8bf*)(p);
  r.h[1] = *(const v8bf*)(p + 16);
  return r.v;
}

// A-fragment (16x32 bf16) from an LDS row: lane holds K = kb..kb+7 and
// kb+16..kb+23 (kb = k*32 + 8*(lane>=16)), matching VGPR0-3 / VGPR4-7 layout.
__device__ inline v16bf load_afrag(const bf16* __restrict__ row, int kb) {
  V16 r;
  r.h[0] = *(const v8bf*)(row + kb);
  r.h[1] = *(const v8bf*)(row + kb + 16);
  return r.v;
}

// Branch-free activations: pure v_exp_f32 / v_rcp_f32 / FMA so the VALU work
// can co-execute with multicycle WMMA (XDL) ops instead of exec-mask branching.
__device__ inline float fast_sigmoid(float x) {
  return __builtin_amdgcn_rcpf(1.0f + __expf(-x));
}
__device__ inline float fast_tanh(float x) {
  return 2.0f * __builtin_amdgcn_rcpf(1.0f + __expf(-2.0f * x)) - 1.0f;
}

// ---------------------------------------------------------------------------
// Kernel 0: weight fp32->bf16 conversion + state zeroing.
// id ranges: Win[0,131072) Wg[131072,262144) W1[262144,524288)
//            W2[524288,655360) state-zero[655360,688128)
// ---------------------------------------------------------------------------
__global__ __launch_bounds__(256) void s4m_prep(
    const float* __restrict__ Win, const float* __restrict__ Wg,
    const float* __restrict__ W1,  const float* __restrict__ W2,
    bf16* __restrict__ Winb, bf16* __restrict__ Wgb,
    bf16* __restrict__ W1b,  bf16* __restrict__ W2b,
    float* __restrict__ state)
{
  int i = blockIdx.x * 256 + threadIdx.x;
  if      (i < 131072) Winb[i]          = (bf16)Win[i];
  else if (i < 262144) Wgb[i - 131072]  = (bf16)Wg[i - 131072];
  else if (i < 524288) W1b[i - 262144]  = (bf16)W1[i - 262144];
  else if (i < 655360) W2b[i - 524288]  = (bf16)W2[i - 524288];
  else if (i < 688128) state[i - 655360] = 0.0f;
}

// ---------------------------------------------------------------------------
// Kernel 1: fused input GEMMs + activations + diagonal scan.
// One block per (batch b, time-chunk c of 256 steps). 256 threads = 8 waves.
// ---------------------------------------------------------------------------
__global__ __launch_bounds__(256) void s4m_main(
    const float* __restrict__ x,
    const bf16* __restrict__ Winb, const float* __restrict__ b_in,
    const bf16* __restrict__ Wgb,  const float* __restrict__ b_g,
    const float* __restrict__ logit_decay,
    float* __restrict__ state)
{
  extern __shared__ bf16 smem[];              // CHUNK * LPITCH bf16 = 136 KB
  const int tid = threadIdx.x;
  const int c   = blockIdx.x % NCHUNK;        // time chunk
  const int b   = blockIdx.x / NCHUNK;        // batch

  // --- Stage x[b, c*256 : c*256+256, 0:256] fp32 -> bf16 into LDS ----------
  const float4* xg = (const float4*)(x + ((size_t)b * T_ + (size_t)c * CHUNK) * D_);
  for (int it = 0; it < 64; ++it) {           // 16384 float4 / 256 threads
    int i = it * 256 + tid;                   // coalesced across lanes
    float4 f = xg[i];
    int row = i >> 6;                         // 64 float4 per 256-wide row
    int c4  = i & 63;
    v4bf p;
    p[0] = (bf16)f.x; p[1] = (bf16)f.y; p[2] = (bf16)f.z; p[3] = (bf16)f.w;
    *(v4bf*)(smem + row * LPITCH + c4 * 4) = p;
  }
  __syncthreads();

  const int lane = tid & 31;
  const int wv   = tid >> 5;                  // wave 0..7
  const int col  = lane & 15;                 // N column / A row within tile
  const int hi   = lane >> 4;                 // half-wave select
  const int kA0  = hi * 8;                    // A K-offset per ISA layout
  const int kB0  = hi * 8;                    // B K-offset per ISA layout

  // Each wave owns 4 N-tiles of S (8 waves * 4 * 16 = 512 columns).
  for (int i = 0; i < 4; ++i) {
    const int sCol = (wv * 4 + i) * 16 + col;

    // Hoist all 8 K-step B fragments for both weight matrices (registers).
    v16bf Bin[8], Bg[8];
#pragma unroll
    for (int k = 0; k < 8; ++k) {
      Bin[k] = load_bfrag(Winb, sCol, D_, k * 32 + kB0);
      Bg[k]  = load_bfrag(Wgb,  sCol, D_, k * 32 + kB0);
    }
    const float bu = b_in[sCol];
    const float bg = b_g[sCol];
    const float d  = fast_sigmoid(logit_decay[sCol]);
    const float d2 = d * d, d4 = d2 * d2, d8 = d4 * d4, d16 = d8 * d8;

    float st = 0.0f;                          // chunk-local scan state (per col)
#pragma unroll 1
    for (int m = 0; m < 16; ++m) {            // 16 M-tiles of 16 timesteps
      v8f accU = {}; v8f accG = {};
      const bf16* arow = smem + (m * 16 + col) * LPITCH;
#pragma unroll
      for (int k = 0; k < 8; ++k) {           // K = 256 / 32
        v16bf A = load_afrag(arow, k * 32 + kA0);
        accU = __builtin_amdgcn_wmma_f32_16x16x32_bf16(
            false, A, false, Bin[k], (short)0, accU, false, false);
        accG = __builtin_amdgcn_wmma_f32_16x16x32_bf16(
            false, A, false, Bg[k], (short)0, accG, false, false);
      }
      // Activations + intra-tile Horner scan over this lane's 8 timesteps.
      // C-tile layout: lanes<16 hold t = m*16 + r, lanes>=16 hold t = m*16+8+r.
      float tp = 0.0f;
#pragma unroll
      for (int r = 0; r < 8; ++r) {
        float uv = fast_tanh(accU[r] + bu);
        float gv = fast_sigmoid(accG[r] + bg);
        tp = tp * d + gv * uv;                // sum_r d^(7-r) * gu_r
      }
      if (hi == 0) tp *= d8;                  // low half = earlier timesteps
      float tilesum = tp + __shfl_xor(tp, 16);
      st = st * d16 + tilesum;                // tile-level Horner over m
    }
    // Exact cross-chunk combine: weight this chunk by d^(256*(7-c)).
    float cw = __powf(d, (float)(CHUNK * (NCHUNK - 1 - c)));
    if (hi == 0) atomicAdd(&state[b * S_ + sCol], st * cw);
  }
}

// ---------------------------------------------------------------------------
// Kernel 2: head. 4 blocks, each handles 16 batch rows.
// LN -> [16x512]@W1^T + GELU(exact) -> [16x512]@W2^T + b2 -> out[16x256].
// ---------------------------------------------------------------------------
__global__ __launch_bounds__(256) void s4m_head(
    const float* __restrict__ state,
    const float* __restrict__ ln_g, const float* __restrict__ ln_b,
    const bf16* __restrict__ W1b, const float* __restrict__ b1,
    const bf16* __restrict__ W2b, const float* __restrict__ b2,
    float* __restrict__ out)
{
  __shared__ bf16 hbuf[16 * HPITCH];
  __shared__ bf16 h2buf[16 * HPITCH];
  const int tid = threadIdx.x;
  const int b0  = blockIdx.x * 16;

  // --- LayerNorm: thread (r = tid>>4, j = tid&15), 32 cols each ------------
  {
    const int r = tid >> 4, j = tid & 15;
    const float* row = state + (size_t)(b0 + r) * S_;
    float s = 0.f, sq = 0.f;
#pragma unroll
    for (int k2 = 0; k2 < 32; ++k2) {
      float v = row[j + k2 * 16];
      s += v; sq += v * v;
    }
#pragma unroll
    for (int msk = 8; msk >= 1; msk >>= 1) {  // reduce within 16-lane group
      s  += __shfl_xor(s,  msk);
      sq += __shfl_xor(sq, msk);
    }
    float mean = s * (1.0f / S_);
    float var  = sq * (1.0f / S_) - mean * mean;
    float inv  = rsqrtf(var + 1e-5f);
#pragma unroll
    for (int k2 = 0; k2 < 32; ++k2) {
      int cidx = j + k2 * 16;
      float v = (row[cidx] - mean) * inv * ln_g[cidx] + ln_b[cidx];
      hbuf[r * HPITCH + cidx] = (bf16)v;
    }
  }
  __syncthreads();

  const int lane = tid & 31, wv = tid >> 5;
  const int col = lane & 15, hi = lane >> 4;
  const int kA0 = hi * 8, kB0 = hi * 8;

  // --- GEMM1: [16x512] x W1^T (512x512), exact GELU -> h2buf ---------------
  for (int i = 0; i < 4; ++i) {
    const int sCol = (wv * 4 + i) * 16 + col;
    v8f acc = {};
#pragma unroll
    for (int k = 0; k < 16; ++k) {
      v16bf A  = load_afrag(hbuf + col * HPITCH, k * 32 + kA0);
      v16bf Bf = load_bfrag(W1b, sCol, S_, k * 32 + kB0);
      acc = __builtin_amdgcn_wmma_f32_16x16x32_bf16(
          false, A, false, Bf, (short)0, acc, false, false);
    }
    float bb = b1[sCol];
#pragma unroll
    for (int r = 0; r < 8; ++r) {
      float v = acc[r] + bb;
      v = 0.5f * v * (1.0f + erff(v * 0.70710678118f));   // exact GELU
      h2buf[(r + hi * 8) * HPITCH + sCol] = (bf16)v;
    }
  }
  __syncthreads();

  // --- GEMM2: [16x512] x W2^T (256x512) + b2 -> out ------------------------
  for (int i = 0; i < 2; ++i) {
    const int oCol = (wv * 2 + i) * 16 + col;
    v8f acc = {};
#pragma unroll
    for (int k = 0; k < 16; ++k) {
      v16bf A  = load_afrag(h2buf + col * HPITCH, k * 32 + kA0);
      v16bf Bf = load_bfrag(W2b, oCol, S_, k * 32 + kB0);
      acc = __builtin_amdgcn_wmma_f32_16x16x32_bf16(
          false, A, false, Bf, (short)0, acc, false, false);
    }
    float bb = b2[oCol];
#pragma unroll
    for (int r = 0; r < 8; ++r) {
      out[(size_t)(b0 + r + hi * 8) * O_ + oCol] = acc[r] + bb;
    }
  }
}

// ---------------------------------------------------------------------------
extern "C" void kernel_launch(void* const* d_in, const int* in_sizes, int n_in,
                              void* d_out, int out_size, void* d_ws, size_t ws_size,
                              hipStream_t stream)
{
  (void)in_sizes; (void)n_in; (void)out_size; (void)ws_size;
  const float* x   = (const float*)d_in[0];
  const float* Win = (const float*)d_in[1];
  const float* bin = (const float*)d_in[2];
  const float* Wg  = (const float*)d_in[3];
  const float* bg  = (const float*)d_in[4];
  const float* ld  = (const float*)d_in[5];
  const float* lng = (const float*)d_in[6];
  const float* lnb = (const float*)d_in[7];
  const float* W1  = (const float*)d_in[8];
  const float* b1  = (const float*)d_in[9];
  const float* W2  = (const float*)d_in[10];
  const float* b2  = (const float*)d_in[11];
  float* out = (float*)d_out;

  // Workspace layout (1,441,792 B total):
  char* ws = (char*)d_ws;
  float* state = (float*)ws;                    //  64*512 f32   = 131072 B
  bf16*  Winb  = (bf16*)(ws + 131072);          // 512*256 bf16  = 262144 B
  bf16*  Wgb   = (bf16*)(ws + 393216);          // 512*256 bf16  = 262144 B
  bf16*  W1b   = (bf16*)(ws + 655360);          // 512*512 bf16  = 524288 B
  bf16*  W2b   = (bf16*)(ws + 1179648);         // 256*512 bf16  = 262144 B

  s4m_prep<<<2688, 256, 0, stream>>>(Win, Wg, W1, W2, Winb, Wgb, W1b, W2b, state);

  const size_t ldsBytes = (size_t)CHUNK * LPITCH * sizeof(bf16);   // 139264 B
  s4m_main<<<B_ * NCHUNK, 256, ldsBytes, stream>>>(x, Winb, bin, Wgb, bg, ld, state);

  s4m_head<<<B_ / 16, 256, 0, stream>>>(state, lng, lnb, W1b, b1, W2b, b2, out);
}
